// QLSTM_65481071408636
// MI455X (gfx1250) — matrix operand order repaired and checked by reading, not compiled, not run on hardware
//
#include <hip/hip_runtime.h>
#include <hip/hip_bf16.h>

// ---------------------------------------------------------------------------
// LayerNorm-LSTM, persistent-kernel design for MI455X (gfx1250, wave32).
//   T=1024 steps, B=64, D=H=512, DH=1024, 4H=2048 gate columns.
//   32 persistent workgroups; WG k owns h-columns [16k,16k+16) and the four
//   matching 16-wide gate slices (64 gate columns). Its 64x1024 bf16 weight
//   slice lives in LDS for the whole kernel (128 KB of the 320 KB WGP LDS),
//   DMA'd once via global_load_async_to_lds_b128. Per step: layernorm ->
//   bf16 A in LDS -> fully-unrolled double-buffered WMMA GEMM -> gates
//   (v_tanh_f32) -> grid barrier. h state ping-pongs in global workspace.
// ---------------------------------------------------------------------------

#define T_STEPS 1024
#define BATCH   64
#define DIM     512
#define HID     512
#define DH      1024
#define GATES4  2048
#define NWG     32
#define NTHR    256
#define HCOLS   16
#define GCOLS   64
#define LN_EPS  1e-5f

typedef __attribute__((ext_vector_type(16))) __bf16 v16bf;
typedef __attribute__((ext_vector_type(8)))  __bf16 v8bf;
typedef __attribute__((ext_vector_type(8)))  float  v8f;

// ---- dynamic LDS layout (bytes) -------------------------------------------
#define LDS_WT    0                        // weights: 64 rows x 2048 B (bf16[64][1024])
#define LDS_A     131072                   // A matrix: 64 rows x 2048 B (bf16[64][1024])
#define LDS_Z     262144                   // z tiles:  64 x 64 f32   (16 KB)
#define LDS_CC    (262144 + 16384)         // c state:  64 x 16 f32   (4 KB)
#define LDS_BIAS  (262144 + 16384 + 4096)  // 64 f32
#define SMEM_BYTES (262144 + 16384 + 4096 + 256)   // ~283 KB of the 320 KB LDS

// ---------------------------------------------------------------------------
__global__ void qlstm_init(float* hbuf2, unsigned* bar) {
  unsigned i = blockIdx.x * blockDim.x + threadIdx.x;   // 256*256 = 65536 = 2*B*H
  hbuf2[i] = 0.0f;
  if (i == 0) { bar[0] = 0u; bar[1] = 0u; }
}

// Transpose + convert the four [1024,512] f32 gate weights into an N-major
// bf16 buffer Wt[n][k], n in [0,2048), so the persistent kernel can async-DMA
// contiguous 32 KB slices straight into LDS.
__global__ void qlstm_prep_weights(const float* __restrict__ Wf,
                                   const float* __restrict__ Wi,
                                   const float* __restrict__ Wg,
                                   const float* __restrict__ Wo,
                                   __bf16* __restrict__ Wt) {
  int n = blockIdx.x;                                    // 0..2047
  const float* W = (n < 512) ? Wf : (n < 1024) ? Wi : (n < 1536) ? Wg : Wo;
  int hcol = n & 511;
  __bf16* dst = Wt + (size_t)n * DH;
  for (int k = threadIdx.x; k < DH; k += NTHR)
    dst[k] = (__bf16)W[k * HID + hcol];
}

// ---------------------------------------------------------------------------
__device__ __forceinline__ void grid_barrier(unsigned* bar) {
  __syncthreads();
  if (threadIdx.x == 0) {
    __threadfence();
    unsigned gen = __hip_atomic_load(&bar[1], __ATOMIC_RELAXED, __HIP_MEMORY_SCOPE_AGENT);
    unsigned arr = __hip_atomic_fetch_add(&bar[0], 1u, __ATOMIC_ACQ_REL, __HIP_MEMORY_SCOPE_AGENT);
    if (arr == NWG - 1) {
      __hip_atomic_store(&bar[0], 0u, __ATOMIC_RELAXED, __HIP_MEMORY_SCOPE_AGENT);
      __hip_atomic_fetch_add(&bar[1], 1u, __ATOMIC_RELEASE, __HIP_MEMORY_SCOPE_AGENT);
    } else {
      while (__hip_atomic_load(&bar[1], __ATOMIC_ACQUIRE, __HIP_MEMORY_SCOPE_AGENT) == gen)
        __builtin_amdgcn_s_sleep(1);
    }
  }
  __syncthreads();
}

// CDNA5 hardware tanh (TRANS pipe). One independent op (v_nop) required
// before the result can be consumed; the compiler can't see through the asm,
// so we bundle it.
__device__ __forceinline__ float tanh_hw(float x) {
  float r;
  asm volatile("v_tanh_f32 %0, %1\n\tv_nop" : "=v"(r) : "v"(x));
  return r;
}
__device__ __forceinline__ float sigmoid_hw(float x) {
  return __builtin_fmaf(0.5f, tanh_hw(0.5f * x), 0.5f);   // 0.5*(1+tanh(x/2))
}

// A-fragment (16x32 bf16, WMMA layout, ISA 7.12.2): lane L holds row M=L&15,
// K = (L>>4)*8 + {0..7} (lo 16B) and K = 16 + (L>>4)*8 + {0..7} (hi 16B).
__device__ __forceinline__ v16bf ld_afrag(const char* ldsA, int mtile, int kt, int lane) {
  int row   = mtile * 16 + (lane & 15);
  int halfk = lane >> 4;
  const char* p = ldsA + row * 2048 + kt * 64 + halfk * 16;
  union { struct { v8bf lo, hi; } s; v16bf v; } u;
  u.s.lo = *(const v8bf*)p;
  u.s.hi = *(const v8bf*)(p + 32);
  return u.v;
}

// B-fragment (32x16 bf16): lane L holds column N=L&15, K = (L>>4)*8 + {0..7}
// (lo 16B) and K = 16 + (L>>4)*8 + {0..7} (hi 16B) of the N-major weight slice.
__device__ __forceinline__ v16bf ld_bfrag(const char* ldsWt, int ntile, int kt, int lane) {
  int n     = ntile * 16 + (lane & 15);
  int halfk = lane >> 4;
  const char* p = ldsWt + n * 2048 + kt * 64 + halfk * 16;
  union { struct { v8bf lo, hi; } s; v16bf v; } u;
  u.s.lo = *(const v8bf*)p;
  u.s.hi = *(const v8bf*)(p + 32);
  return u.v;
}

// CDNA5 async copy: 16 bytes per lane, global -> LDS, tracked by ASYNCcnt.
__device__ __forceinline__ void async_copy_b128(unsigned lds_addr, unsigned long long gaddr) {
  asm volatile("global_load_async_to_lds_b128 %0, %1, off"
               :: "v"(lds_addr), "v"(gaddr) : "memory");
}
__device__ __forceinline__ void wait_asynccnt0() {
  asm volatile("s_wait_asynccnt 0" ::: "memory");
}

// ---------------------------------------------------------------------------
__global__ void __launch_bounds__(NTHR, 1)
qlstm_persistent(const float* __restrict__ inputs,   // [T,B,D]
                 const float* __restrict__ ln_g,     // [DH]
                 const float* __restrict__ ln_b,     // [DH]
                 const float* __restrict__ bfv, const float* __restrict__ biv,
                 const float* __restrict__ bgv, const float* __restrict__ bov,
                 const __bf16* __restrict__ Wt,      // [2048][1024] bf16, N-major
                 float* __restrict__ hbuf2,          // 2 x [B][H] ping-pong
                 unsigned* __restrict__ bar,
                 float* __restrict__ out) {          // [T*B*H + B*H + B*H]
  extern __shared__ char smem[];
  char*  ldsWt  = smem + LDS_WT;
  char*  ldsA   = smem + LDS_A;
  float* ldsZ   = (float*)(smem + LDS_Z);
  float* ldsC   = (float*)(smem + LDS_CC);
  float* ldsBias= (float*)(smem + LDS_BIAS);

  const int wg   = blockIdx.x;
  const int tid  = threadIdx.x;
  const int lane = tid & 31;
  const int wave = tid >> 5;

  // ---- one-time: DMA the 64x1024 bf16 weight slice into LDS (4 x 32 KB) ---
  for (int g = 0; g < 4; ++g) {
    const char* src = (const char*)(Wt + ((size_t)(g * 512 + wg * HCOLS)) * DH);
    unsigned ldsbase = (unsigned)(size_t)(ldsWt + g * 16 * 2048);
    #pragma unroll
    for (int it = 0; it < 8; ++it) {
      unsigned off = (unsigned)(it * NTHR + tid) * 16u;
      async_copy_b128(ldsbase + off, (unsigned long long)src + off);
    }
  }
  // biases for this WG's 64 gate columns
  if (tid < GCOLS) {
    int g = tid >> 4;
    int col = wg * HCOLS + (tid & 15);
    const float* bp = (g == 0) ? bfv : (g == 1) ? biv : (g == 2) ? bgv : bov;
    ldsBias[tid] = bp[col];
  }
  // zero cell state
  for (int j = tid; j < BATCH * HCOLS; j += NTHR) ldsC[j] = 0.0f;
  wait_asynccnt0();
  __syncthreads();

  const int row = tid >> 2;          // 0..63 : batch row (4 threads per row)
  const int q   = tid & 3;           // column-quad interleave

  // GEMM tile assignment + per-lane bias (C/D layout: bias depends on N only)
  const int mtile = wave & 3;
  const int nb    = (wave >> 2) * 2;         // ntile pair {nb, nb+1}
  const float bias0 = ldsBias[nb * 16 + (lane & 15)];
  const float bias1 = ldsBias[(nb + 1) * 16 + (lane & 15)];

  const size_t TBH = (size_t)T_STEPS * BATCH * HID;

  for (int t = 0; t < T_STEPS; ++t) {
    const float* x_t   = inputs + (size_t)t * BATCH * DIM;
    const float* hread = hbuf2 + (size_t)(t & 1) * BATCH * HID;
    float*       hwrit = hbuf2 + (size_t)((t + 1) & 1) * BATCH * HID;

    // prefetch next timestep's x into GL2 (global_prefetch_b8)
    if (t + 1 < T_STEPS)
      __builtin_prefetch(inputs + (size_t)(t + 1) * BATCH * DIM + tid * 128, 0, 3);

    // ---- pass 1: mean / var over comb=[x_t|h] (1024 cols per row) ---------
    // 4 threads per row; thread q owns column-quads qd = q + 4c.
    const float4* px4 = (const float4*)(x_t   + row * DIM);
    const float4* ph4 = (const float4*)(hread + row * HID);
    float sum = 0.0f, sq = 0.0f;
    #pragma unroll 4
    for (int c = 0; c < 32; ++c) {             // x half: quads 0..127
      float4 v = px4[q + 4 * c];
      sum += (v.x + v.y) + (v.z + v.w);
      sq  += (v.x * v.x + v.y * v.y) + (v.z * v.z + v.w * v.w);
    }
    #pragma unroll 4
    for (int c = 0; c < 32; ++c) {             // h half: quads 128..255
      float4 v = ph4[q + 4 * c];
      sum += (v.x + v.y) + (v.z + v.w);
      sq  += (v.x * v.x + v.y * v.y) + (v.z * v.z + v.w * v.w);
    }
    sum += __shfl_xor(sum, 1); sum += __shfl_xor(sum, 2);
    sq  += __shfl_xor(sq, 1);  sq  += __shfl_xor(sq, 2);
    float mean = sum * (1.0f / DH);
    float var  = sq * (1.0f / DH) - mean * mean;
    float rstd = rsqrtf(var + LN_EPS);

    // ---- pass 2: normalize, scale, convert to bf16, write A into LDS ------
    const float4* lg4 = (const float4*)ln_g;
    const float4* lb4 = (const float4*)ln_b;
    #pragma unroll 2
    for (int half = 0; half < 2; ++half) {
      const float4* src4 = half ? ph4 : px4;
      #pragma unroll 4
      for (int c = 0; c < 32; ++c) {
        int qd  = q + 4 * c;                   // quad within this half
        int gqd = half * 128 + qd;             // global quad 0..255
        float4 v = src4[qd];
        float4 g4 = lg4[gqd];
        float4 b4 = lb4[gqd];
        union { __bf16 h[4]; unsigned long long u; } pk;
        pk.h[0] = (__bf16)((v.x - mean) * rstd * g4.x + b4.x);
        pk.h[1] = (__bf16)((v.y - mean) * rstd * g4.y + b4.y);
        pk.h[2] = (__bf16)((v.z - mean) * rstd * g4.z + b4.z);
        pk.h[3] = (__bf16)((v.w - mean) * rstd * g4.w + b4.w);
        *(unsigned long long*)(ldsA + row * 2048 + gqd * 8) = pk.u;
      }
    }
    __syncthreads();

    // ---- GEMM: [64,1024] x [1024,64] via v_wmma_f32_16x16x32_bf16 ---------
    // FULLY unrolled K loop with a double-buffered prefetch: buffer indices
    // are compile-time so the rotation is pure SSA renaming (no v_movs), DS
    // offsets fold into the ds_load_b128 immediate, and loads for kt+1 sit
    // between the kt WMMAs and any overwrite of their source registers.
    {
      v8f acc0, acc1;
      #pragma unroll
      for (int j = 0; j < 8; ++j) { acc0[j] = bias0; acc1[j] = bias1; }

      v16bf aB[2], b0B[2], b1B[2];
      aB[0]  = ld_afrag(ldsA,  mtile,  0, lane);
      b0B[0] = ld_bfrag(ldsWt, nb,     0, lane);
      b1B[0] = ld_bfrag(ldsWt, nb + 1, 0, lane);
      #pragma unroll
      for (int kt = 0; kt < 32; ++kt) {
        const int cur = kt & 1, nxt = cur ^ 1;
        if (kt < 31) {
          aB[nxt]  = ld_afrag(ldsA,  mtile,  kt + 1, lane);
          b0B[nxt] = ld_bfrag(ldsWt, nb,     kt + 1, lane);
          b1B[nxt] = ld_bfrag(ldsWt, nb + 1, kt + 1, lane);
        }
        acc0 = __builtin_amdgcn_wmma_f32_16x16x32_bf16(false, aB[cur], false, b0B[cur],
                                                       (short)0, acc0, false, false);
        acc1 = __builtin_amdgcn_wmma_f32_16x16x32_bf16(false, aB[cur], false, b1B[cur],
                                                       (short)0, acc1, false, false);
      }
      // C/D layout: VGPR j, lane L -> (M = j + 8*(L>>4), N = L&15)
      int rbase = mtile * 16 + 8 * (lane >> 4);
      int cn    = lane & 15;
      #pragma unroll
      for (int j = 0; j < 8; ++j) {
        ldsZ[(rbase + j) * GCOLS + nb * 16 + cn]       = acc0[j];
        ldsZ[(rbase + j) * GCOLS + (nb + 1) * 16 + cn] = acc1[j];
      }
    }
    __syncthreads();

    // ---- gates, cell update, residual; write h ----------------------------
    // (bias already folded into the accumulators)
    #pragma unroll
    for (int jj = 0; jj < 4; ++jj) {
      int e   = tid + NTHR * jj;          // 0..1023 over [64 rows][16 cols]
      int r   = e >> 4;
      int col = e & 15;
      int gcol = wg * HCOLS + col;
      float zf = ldsZ[r * GCOLS +  0 + col];
      float zi = ldsZ[r * GCOLS + 16 + col];
      float zg = ldsZ[r * GCOLS + 32 + col];
      float zo = ldsZ[r * GCOLS + 48 + col];
      float f = sigmoid_hw(zf);
      float i = sigmoid_hw(zi);
      float g = tanh_hw(zg);
      float o = sigmoid_hw(zo);
      float c = f * ldsC[e] + i * g;
      ldsC[e] = c;
      float h = o * tanh_hw(c) + x_t[r * DIM + gcol];   // residual add
      hwrit[r * HID + gcol] = h;
      out[((size_t)t * BATCH + r) * HID + gcol] = h;
      if (t == T_STEPS - 1) {
        out[TBH + r * HID + gcol]               = h;  // hx
        out[TBH + BATCH * HID + r * HID + gcol] = c;  // cx
      }
    }

    grid_barrier(bar);   // everyone done writing h before next step reads it
  }
}

// ---------------------------------------------------------------------------
extern "C" void kernel_launch(void* const* d_in, const int* in_sizes, int n_in,
                              void* d_out, int out_size, void* d_ws, size_t ws_size,
                              hipStream_t stream) {
  const float* inputs = (const float*)d_in[0];
  const float* ln_g   = (const float*)d_in[1];
  const float* ln_b   = (const float*)d_in[2];
  const float* Wf     = (const float*)d_in[3];
  const float* bfv    = (const float*)d_in[4];
  const float* Wi     = (const float*)d_in[5];
  const float* biv    = (const float*)d_in[6];
  const float* Wg     = (const float*)d_in[7];
  const float* bgv    = (const float*)d_in[8];
  const float* Wo     = (const float*)d_in[9];
  const float* bov    = (const float*)d_in[10];
  float* out = (float*)d_out;

  // workspace layout
  unsigned* bar  = (unsigned*)d_ws;                                  // 2 u32 (+pad)
  float*    hbuf = (float*)((char*)d_ws + 1024);                     // 2*B*H f32
  __bf16*   Wt   = (__bf16*)((char*)d_ws + 1024 + 2 * BATCH * HID * 4); // 2048*1024 bf16

  static_assert(SMEM_BYTES <= 320 * 1024, "exceeds WGP LDS");
  hipFuncSetAttribute((const void*)qlstm_persistent,
                      hipFuncAttributeMaxDynamicSharedMemorySize, SMEM_BYTES);

  qlstm_init<<<256, 256, 0, stream>>>(hbuf, bar);
  qlstm_prep_weights<<<GATES4, NTHR, 0, stream>>>(Wf, Wi, Wg, Wo, Wt);
  qlstm_persistent<<<NWG, NTHR, SMEM_BYTES, stream>>>(
      inputs, ln_g, ln_b, bfv, biv, bgv, bov, Wt, hbuf, bar, out);
}
